// ReckoningLayer_23124103922106
// MI455X (gfx1250) — matrix operand order repaired and checked by reading, not compiled, not run on hardware
//
#include <hip/hip_runtime.h>
#include <stdint.h>

// ---------------------------------------------------------------------------
// ReckoningLayer forward for MI455X (gfx1250, wave32, WMMA).
// ternary_ste(w) forward == w exactly, so all GEMMs use the raw weights,
// cast to f16 with fp32 WMMA accumulation (v_wmma_f32_16x16x32_f16).
// GEMMs: 128x128 block tile staged through LDS with double-buffered
// GLOBAL_LOAD_ASYNC_TO_LDS_B128 (ASYNCcnt), fragments fed by ds_load_b128.
// ---------------------------------------------------------------------------

typedef _Float16 v16h __attribute__((ext_vector_type(16)));
typedef _Float16 v8h  __attribute__((ext_vector_type(8)));
typedef float    v8f  __attribute__((ext_vector_type(8)));

#define B_      4
#define T_      2048
#define D_      1024
#define BT_     8192
#define NC_     4096
#define DC_     256
#define TOPK_   8
#define DFF_    4096
#define NCHUNK_ 16
#define CHLEN_  128
#define PAD_    40   // 32-half K-slice padded to 40 halfs (80B = 20 banks;
                     // 16 lanes * 20 banks mod 64 covers all banks once)

// --- CDNA5 async global->LDS copy helpers (ASYNCcnt) -----------------------
__device__ __forceinline__ uint32_t lds_off(const void* p) {
  // generic pointers to LDS carry the byte offset in the low 32 bits
  return (uint32_t)(uintptr_t)p;
}
__device__ __forceinline__ void async_b128(uint32_t lds, const _Float16* g) {
  asm volatile("global_load_async_to_lds_b128 %0, %1, off"
               :: "v"(lds), "v"(g) : "memory");
}
__device__ __forceinline__ void wait_async_le4() {
  asm volatile("s_wait_asynccnt 0x4" ::: "memory");
}
__device__ __forceinline__ void wait_async_0() {
  asm volatile("s_wait_asynccnt 0x0" ::: "memory");
}

// ---------------------------------------------------------------------------
// fp32 -> fp16 convert (grid-stride)
// ---------------------------------------------------------------------------
__global__ void convert_f16_kernel(const float* __restrict__ in,
                                   _Float16* __restrict__ out, long n) {
  long i = (long)blockIdx.x * blockDim.x + threadIdx.x;
  long stride = (long)gridDim.x * blockDim.x;
  for (; i < n; i += stride) out[i] = (_Float16)in[i];
}

// ---------------------------------------------------------------------------
// fp32 [R,C] -> fp16 [C,R] tiled transpose-convert
// ---------------------------------------------------------------------------
__global__ void transpose_f16_kernel(const float* __restrict__ in,
                                     _Float16* __restrict__ out, int R, int C) {
  __shared__ float tile[32][33];
  const int tx = threadIdx.x, ty = threadIdx.y;      // block (32,8)
  const int c0 = blockIdx.x * 32, r0 = blockIdx.y * 32;
#pragma unroll
  for (int i = 0; i < 32; i += 8) {
    int r = r0 + ty + i, c = c0 + tx;
    if (r < R && c < C) tile[ty + i][tx] = in[(size_t)r * C + c];
  }
  __syncthreads();
#pragma unroll
  for (int i = 0; i < 32; i += 8) {
    int orow = c0 + ty + i, oc = r0 + tx;
    if (orow < C && oc < R) out[(size_t)orow * R + oc] = (_Float16)tile[tx][ty + i];
  }
}

// ---------------------------------------------------------------------------
// WMMA GEMM: C[M,N] = scale * (A[M,K] @ Bt[N,K]^T) (+ add[M,N])
// A,Bt fp16 row-major. Block: 256 thr = 8 waves; block tile 128(M) x 128(N);
// wave tile 64x32 (4x2 wmma). K staged 32-wide through double-buffered LDS
// via async copies. Requires M%128==0, N%128==0, K%32==0 (all call sites ok).
// ---------------------------------------------------------------------------
template <bool OUT_F16, bool HAS_ADD>
__global__ void gemm_wmma_kernel(const _Float16* __restrict__ A,
                                 const _Float16* __restrict__ Bt,
                                 void* __restrict__ Cout,
                                 const float* __restrict__ add,
                                 int M, int N, int K, float scale) {
  __shared__ _Float16 smA[2][128 * PAD_];
  __shared__ _Float16 smB[2][128 * PAD_];

  const int tid  = threadIdx.x;
  const int wave = tid >> 5;
  const int lane = tid & 31;
  const int half = lane >> 4;    // lanes 0-15 vs 16-31 hold different K chunks
  const int l16  = lane & 15;
  const int blkM = blockIdx.y * 128;
  const int blkN = blockIdx.x * 128;
  const int m0w  = (wave & 1) * 64;   // wave's M offset inside block tile
  const int n0w  = (wave >> 1) * 32;  // wave's N offset inside block tile

  // staging: threads 0-127 copy A row `tid`, threads 128-255 copy B row
  const bool isB = tid >= 128;
  const int  row = tid & 127;
  const _Float16* gsrc = isB ? (Bt + (size_t)(blkN + row) * K)
                             : (A  + (size_t)(blkM + row) * K);
  uint32_t lrow0 = lds_off((isB ? smB[0] : smA[0]) + row * PAD_);
  uint32_t lrow1 = lds_off((isB ? smB[1] : smA[1]) + row * PAD_);

  v8f acc[4][2] = {};
  const int nstage = K >> 5;

  // prologue: stage 0 into buffer 0 (4 x b128 = one 64B K-slice per row)
  {
    const _Float16* g = gsrc;
    async_b128(lrow0, g);          async_b128(lrow0 + 16, g + 8);
    async_b128(lrow0 + 32, g + 16); async_b128(lrow0 + 48, g + 24);
  }

  for (int s = 0; s < nstage; ++s) {
    if (s + 1 < nstage) {           // issue next stage into the other buffer
      uint32_t l = ((s + 1) & 1) ? lrow1 : lrow0;
      const _Float16* g = gsrc + ((s + 1) << 5);
      async_b128(l, g);           async_b128(l + 16, g + 8);
      async_b128(l + 32, g + 16); async_b128(l + 48, g + 24);
      wait_async_le4();             // wait for stage s, keep s+1 in flight
    } else {
      wait_async_0();
    }
    __syncthreads();                // stage s visible to all waves

    const _Float16* la = smA[s & 1];
    const _Float16* lb = smB[s & 1];

    v16h af[4], bf[2];
#pragma unroll
    for (int mt = 0; mt < 4; ++mt) {
      // A frag per lane: halfs [half*8 .. +7] and [16+half*8 .. +7]
      const _Float16* p = la + (m0w + mt * 16 + l16) * PAD_ + half * 8;
      v8h lo = *(const v8h*)p;
      v8h hi = *(const v8h*)(p + 16);
      af[mt] = __builtin_shufflevector(lo, hi, 0,1,2,3,4,5,6,7,8,9,10,11,12,13,14,15);
    }
#pragma unroll
    for (int nt = 0; nt < 2; ++nt) {
      // B frag per lane: halfs [half*8 .. +7] and [16+half*8 .. +7]
      const _Float16* p = lb + (n0w + nt * 16 + l16) * PAD_ + half * 8;
      v8h lo = *(const v8h*)p;
      v8h hi = *(const v8h*)(p + 16);
      bf[nt] = __builtin_shufflevector(lo, hi, 0,1,2,3,4,5,6,7,8,9,10,11,12,13,14,15);
    }
#pragma unroll
    for (int mt = 0; mt < 4; ++mt)
#pragma unroll
      for (int nt = 0; nt < 2; ++nt)
        acc[mt][nt] = __builtin_amdgcn_wmma_f32_16x16x32_f16(
            false, af[mt], false, bf[nt], (short)0, acc[mt][nt], false, false);

    __syncthreads();                // all reads of buffer (s&1) done
  }

  // C/D layout: VGPR r -> row + half*8 + r, col + l16
#pragma unroll
  for (int mt = 0; mt < 4; ++mt)
#pragma unroll
    for (int nt = 0; nt < 2; ++nt)
#pragma unroll
      for (int r = 0; r < 8; ++r) {
        int rowc = blkM + m0w + mt * 16 + half * 8 + r;
        int colc = blkN + n0w + nt * 16 + l16;
        size_t idx = (size_t)rowc * N + colc;
        float v = acc[mt][nt][r] * scale;
        if (HAS_ADD) v += add[idx];
        if (OUT_F16) ((_Float16*)Cout)[idx] = (_Float16)v;
        else         ((float*)Cout)[idx] = v;
      }
}

// ---------------------------------------------------------------------------
// Top-8 of 4096 scores per token (one wave32 per token), softmax, weighted
// gather of cells rows; accumulates readout into y (which holds x@W_in).
// ---------------------------------------------------------------------------
__global__ void topk_readout_kernel(const float* __restrict__ scores,
                                    const float* __restrict__ cells,
                                    float* __restrict__ y) {
  const int wave  = threadIdx.x >> 5;
  const int lane  = threadIdx.x & 31;
  const int token = blockIdx.x * 8 + wave;
  if (token >= BT_) return;
  const float* srow = scores + (size_t)token * NC_;

  float bs[8]; int bi[8];
#pragma unroll
  for (int q = 0; q < 8; ++q) { bs[q] = -3.0e38f; bi[q] = 0; }

  // per-lane sorted (desc) top-8 over its 128 strided scores
  for (int j = lane; j < NC_; j += 32) {
    float cv = srow[j]; int ci = j;
#pragma unroll
    for (int q = 0; q < 8; ++q) {
      if (cv > bs[q]) {
        float tv = bs[q]; int ti = bi[q];
        bs[q] = cv; bi[q] = ci; cv = tv; ci = ti;
      }
    }
  }

  // 8 rounds of wave-argmax; winner pops its local list
  float topw[8]; int topid[8];
#pragma unroll
  for (int k = 0; k < 8; ++k) {
    float cand = bs[0];
    float m = cand;
#pragma unroll
    for (int off = 16; off >= 1; off >>= 1) m = fmaxf(m, __shfl_xor(m, off, 32));
    unsigned long long mask = __ballot(cand == m);
    int src = mask ? (int)__builtin_ctzll(mask) : 0;
    int widx = __shfl(bi[0], src, 32);
    topw[k] = m; topid[k] = widx;
    if (lane == src) {
#pragma unroll
      for (int q = 0; q < 7; ++q) { bs[q] = bs[q + 1]; bi[q] = bi[q + 1]; }
      bs[7] = -3.0e38f;
    }
  }

  // softmax over the (descending) top-8
  float w[8]; float se = 0.f;
#pragma unroll
  for (int k = 0; k < 8; ++k) { w[k] = __expf(topw[k] - topw[0]); se += w[k]; }
  const float inv = 1.f / se;

  // readout: each lane owns 8 consecutive of the 256 cell columns
  float acc[8];
#pragma unroll
  for (int j = 0; j < 8; ++j) acc[j] = 0.f;
#pragma unroll
  for (int k = 0; k < 8; ++k) {
    const float* cr = cells + (size_t)topid[k] * DC_ + lane * 8;
    float wk = w[k] * inv;
#pragma unroll
    for (int j = 0; j < 8; ++j) acc[j] += wk * cr[j];
  }
  float* yr = y + (size_t)token * DC_ + lane * 8;
#pragma unroll
  for (int j = 0; j < 8; ++j) yr[j] += acc[j];
}

// ---------------------------------------------------------------------------
// Running state h[t] = decay*h[t-1] + sigmoid(x*gw+gb)*x, chunked 2-pass scan
// ---------------------------------------------------------------------------
__device__ __forceinline__ float decay_from(const float* sdp) {
  float d = 1.f / (1.f + __expf(-sdp[0]));
  return fminf(fmaxf(d, 0.01f), 0.99f);
}

__global__ void scan_pass1_kernel(const float* __restrict__ x,
                                  const float* __restrict__ gw,
                                  const float* __restrict__ gb,
                                  const float* __restrict__ sdp,
                                  float* __restrict__ fch) {
  const int bid = blockIdx.x;                 // 256 blocks = 4(b)*16(chunk)*4(dblk)
  const int d = (bid & 3) * 256 + threadIdx.x;
  const int chunk = (bid >> 2) & 15;
  const int b = bid >> 6;
  const float decay = decay_from(sdp);
  const float gwd = gw[d], gbd = gb[d];
  const float* xr = x + ((size_t)b * T_ + chunk * CHLEN_) * D_ + d;
  float f = 0.f;
  for (int j = 0; j < CHLEN_; ++j) {
    float xv = xr[(size_t)j * D_];
    float g = 1.f / (1.f + __expf(-(xv * gwd + gbd)));
    f = f * decay + g * xv;
  }
  fch[((size_t)b * NCHUNK_ + chunk) * D_ + d] = f;
}

__global__ void scan_pass2_kernel(const float* __restrict__ fch,
                                  const float* __restrict__ sdp,
                                  float* __restrict__ hpr) {
  const int idx = blockIdx.x * 256 + threadIdx.x;  // 16 blocks -> 4096 threads
  const int b = idx >> 10, d = idx & 1023;
  const float decay = decay_from(sdp);
  const float dL = powf(decay, (float)CHLEN_);
  float H = 0.f;
  for (int c = 0; c < NCHUNK_; ++c) {
    size_t o = ((size_t)b * NCHUNK_ + c) * D_ + d;
    hpr[o] = H;
    H = fch[o] + dL * H;
  }
}

__global__ void scan_pass3_kernel(const float* __restrict__ x,
                                  const float* __restrict__ gw,
                                  const float* __restrict__ gb,
                                  const float* __restrict__ sdp,
                                  const float* __restrict__ hpr,
                                  const float* __restrict__ sscale,
                                  float* __restrict__ state) {
  const int bid = blockIdx.x;
  const int d = (bid & 3) * 256 + threadIdx.x;
  const int chunk = (bid >> 2) & 15;
  const int b = bid >> 6;
  const float decay = decay_from(sdp);
  const float gwd = gw[d], gbd = gb[d], ss = sscale[d];
  const float* xr = x + ((size_t)b * T_ + chunk * CHLEN_) * D_ + d;
  float* sr = state + ((size_t)b * T_ + chunk * CHLEN_) * D_ + d;
  float h = hpr[((size_t)b * NCHUNK_ + chunk) * D_ + d];
  for (int j = 0; j < CHLEN_; ++j) {
    float xv = xr[(size_t)j * D_];
    float g = 1.f / (1.f + __expf(-(xv * gwd + gbd)));
    h = h * decay + g * xv;
    sr[(size_t)j * D_] = h * ss;
  }
}

// ---------------------------------------------------------------------------
// z = x + cell_scale*cell_out + state; x2 = rmsnorm(z, norm_w) (f32 out);
// h  = rmsnorm(x2, ff_norm_w) (f16 out). One block (256 thr) per token.
// ---------------------------------------------------------------------------
__global__ void residual_norm_kernel(const float* __restrict__ x,
                                     const float* __restrict__ cell_out,
                                     const float* __restrict__ state,
                                     const float* __restrict__ csp,
                                     const float* __restrict__ norm_w,
                                     const float* __restrict__ ff_norm_w,
                                     float* __restrict__ x2,
                                     _Float16* __restrict__ hb) {
  __shared__ float red[256];
  const int tid = threadIdx.x;
  const size_t base = (size_t)blockIdx.x * D_;
  const float cs = csp[0];

  float z[4]; float p = 0.f;
#pragma unroll
  for (int i = 0; i < 4; ++i) {
    int d = tid + i * 256;
    float v = x[base + d] + cs * cell_out[base + d] + state[base + d];
    z[i] = v; p += v * v;
  }
  red[tid] = p; __syncthreads();
  for (int s = 128; s > 0; s >>= 1) { if (tid < s) red[tid] += red[tid + s]; __syncthreads(); }
  const float rms1 = rsqrtf(red[0] / (float)D_ + 1e-6f);
  __syncthreads();

  float xx[4]; float q = 0.f;
#pragma unroll
  for (int i = 0; i < 4; ++i) {
    int d = tid + i * 256;
    float v = z[i] * rms1 * norm_w[d];
    xx[i] = v; x2[base + d] = v; q += v * v;
  }
  red[tid] = q; __syncthreads();
  for (int s = 128; s > 0; s >>= 1) { if (tid < s) red[tid] += red[tid + s]; __syncthreads(); }
  const float rms2 = rsqrtf(red[0] / (float)D_ + 1e-6f);

#pragma unroll
  for (int i = 0; i < 4; ++i) {
    int d = tid + i * 256;
    hb[base + d] = (_Float16)(xx[i] * rms2 * ff_norm_w[d]);
  }
}

// ---------------------------------------------------------------------------
// sm = silu(g) * u   (f16 in/out, f32 math)
// ---------------------------------------------------------------------------
__global__ void silu_mul_kernel(const _Float16* __restrict__ g,
                                const _Float16* __restrict__ u,
                                _Float16* __restrict__ sm, long n) {
  long i = (long)blockIdx.x * blockDim.x + threadIdx.x;
  long stride = (long)gridDim.x * blockDim.x;
  for (; i < n; i += stride) {
    float gv = (float)g[i];
    float s = gv / (1.f + __expf(-gv));
    sm[i] = (_Float16)(s * (float)u[i]);
  }
}

// ---------------------------------------------------------------------------
// Launch
// ---------------------------------------------------------------------------
extern "C" void kernel_launch(void* const* d_in, const int* in_sizes, int n_in,
                              void* d_out, int out_size, void* d_ws, size_t ws_size,
                              hipStream_t stream) {
  const float* x      = (const float*)d_in[0];
  const float* Wroute = (const float*)d_in[1];
  const float* cells  = (const float*)d_in[2];
  const float* Win    = (const float*)d_in[3];
  const float* Wout   = (const float*)d_in[4];
  const float* sdecay = (const float*)d_in[5];
  const float* gw     = (const float*)d_in[6];
  const float* gb     = (const float*)d_in[7];
  const float* sscale = (const float*)d_in[8];
  const float* cscale = (const float*)d_in[9];
  const float* Wgate  = (const float*)d_in[10];
  const float* Wup    = (const float*)d_in[11];
  const float* Wdown  = (const float*)d_in[12];
  const float* ffnw   = (const float*)d_in[13];
  const float* normw  = (const float*)d_in[14];
  float* out = (float*)d_out;

  // workspace layout (256B aligned)
  char* ws = (char*)d_ws;
  size_t off = 0;
  auto alloc = [&](size_t bytes) { char* p = ws + off; off = (off + bytes + 255) & ~(size_t)255; return p; };
  _Float16* xb  = (_Float16*)alloc((size_t)BT_ * D_ * 2);
  _Float16* wrb = (_Float16*)alloc((size_t)NC_ * D_ * 2);       // [NC,D] (already N,K)
  _Float16* wib = (_Float16*)alloc((size_t)DC_ * D_ * 2);       // W_in^T  [DC,D]
  _Float16* wob = (_Float16*)alloc((size_t)D_ * DC_ * 2);       // W_out^T [D,DC]
  _Float16* wgb = (_Float16*)alloc((size_t)DFF_ * D_ * 2);      // W_gate^T
  _Float16* wub = (_Float16*)alloc((size_t)DFF_ * D_ * 2);      // W_up^T
  _Float16* wdb = (_Float16*)alloc((size_t)D_ * DFF_ * 2);      // W_down^T
  float* scores = (float*)alloc((size_t)BT_ * NC_ * 4);         // reused for g16/u16
  float* y      = (float*)alloc((size_t)BT_ * DC_ * 4);
  _Float16* yb  = (_Float16*)alloc((size_t)BT_ * DC_ * 2);
  float* co     = (float*)alloc((size_t)BT_ * D_ * 4);
  float* fch    = (float*)alloc((size_t)B_ * NCHUNK_ * D_ * 4);
  float* hpr    = (float*)alloc((size_t)B_ * NCHUNK_ * D_ * 4);
  float* st     = (float*)alloc((size_t)BT_ * D_ * 4);
  float* x2     = (float*)alloc((size_t)BT_ * D_ * 4);
  _Float16* hb  = (_Float16*)alloc((size_t)BT_ * D_ * 2);
  _Float16* sm  = (_Float16*)alloc((size_t)BT_ * DFF_ * 2);
  // overlay: scores buffer (128MB f32) is dead after top-k; host g16/u16 there
  _Float16* g16 = (_Float16*)scores;
  _Float16* u16 = (_Float16*)((char*)scores + (size_t)BT_ * DFF_ * 2);

  const dim3 tb(256);
  const dim3 tp(32, 8);

  // weight/activation conversion
  convert_f16_kernel<<<8192, tb, 0, stream>>>(x, xb, (long)BT_ * D_);
  convert_f16_kernel<<<4096, tb, 0, stream>>>(Wroute, wrb, (long)NC_ * D_);
  transpose_f16_kernel<<<dim3(DC_ / 32, D_ / 32), tp, 0, stream>>>(Win, wib, D_, DC_);
  transpose_f16_kernel<<<dim3(D_ / 32, DC_ / 32), tp, 0, stream>>>(Wout, wob, DC_, D_);
  transpose_f16_kernel<<<dim3(DFF_ / 32, D_ / 32), tp, 0, stream>>>(Wgate, wgb, D_, DFF_);
  transpose_f16_kernel<<<dim3(DFF_ / 32, D_ / 32), tp, 0, stream>>>(Wup, wub, D_, DFF_);
  transpose_f16_kernel<<<dim3(D_ / 32, DFF_ / 32), tp, 0, stream>>>(Wdown, wdb, DFF_, D_);

  // scores = x @ W_route^T / 32
  gemm_wmma_kernel<false, false><<<dim3(NC_ / 128, BT_ / 128), tb, 0, stream>>>(
      xb, wrb, (void*)scores, nullptr, BT_, NC_, D_, 0.03125f);
  // y = x @ W_in
  gemm_wmma_kernel<false, false><<<dim3(DC_ / 128, BT_ / 128), tb, 0, stream>>>(
      xb, wib, (void*)y, nullptr, BT_, DC_, D_, 1.0f);
  // y += softmax-topk readout of cells
  topk_readout_kernel<<<BT_ / 8, tb, 0, stream>>>(scores, cells, y);
  convert_f16_kernel<<<2048, tb, 0, stream>>>(y, yb, (long)BT_ * DC_);
  // cell_out = y @ W_out
  gemm_wmma_kernel<false, false><<<dim3(D_ / 128, BT_ / 128), tb, 0, stream>>>(
      yb, wob, (void*)co, nullptr, BT_, D_, DC_, 1.0f);

  // running state (chunked scan)
  scan_pass1_kernel<<<B_ * NCHUNK_ * (D_ / 256), tb, 0, stream>>>(x, gw, gb, sdecay, fch);
  scan_pass2_kernel<<<B_ * (D_ / 256), tb, 0, stream>>>(fch, sdecay, hpr);
  scan_pass3_kernel<<<B_ * NCHUNK_ * (D_ / 256), tb, 0, stream>>>(x, gw, gb, sdecay, hpr, sscale, st);

  // residual + double rmsnorm -> x2 (f32), h (f16)
  residual_norm_kernel<<<BT_, tb, 0, stream>>>(x, co, st, cscale, normw, ffnw, x2, hb);

  // FFN
  gemm_wmma_kernel<true, false><<<dim3(DFF_ / 128, BT_ / 128), tb, 0, stream>>>(
      hb, wgb, (void*)g16, nullptr, BT_, DFF_, D_, 1.0f);
  gemm_wmma_kernel<true, false><<<dim3(DFF_ / 128, BT_ / 128), tb, 0, stream>>>(
      hb, wub, (void*)u16, nullptr, BT_, DFF_, D_, 1.0f);
  silu_mul_kernel<<<8192, tb, 0, stream>>>(g16, u16, sm, (long)BT_ * DFF_);
  // out = x2 + sm @ W_down
  gemm_wmma_kernel<false, true><<<dim3(D_ / 128, BT_ / 128), tb, 0, stream>>>(
      sm, wdb, (void*)out, x2, BT_, D_, DFF_, 1.0f);
}
